// LSHAttention_19756849562257
// MI455X (gfx1250) — compile-verified, matrix operand order checked by reading, not run on hardware
//
#include <hip/hip_runtime.h>
#include <hip/hip_bf16.h>

// ---------------------------------------------------------------------------
// LSH attention for MI455X (gfx1250), wave32 + WMMA bf16, f32 accumulate.
//   B=4, S=2048, E=1024, NUM_HASHES=8.
// Workspace layout (~70 MB):
//   hbf  : [B*S, E]  bf16   hidden downconverted
//   wq/wk/wv : [E, E] bf16  weights downconverted
//   Q    : [B*S, E]  bf16
//   K    : [B*S, E]  bf16
//   Vt   : [B, E, S] bf16   V stored TRANSPOSED for contiguous PV B-frags
// ---------------------------------------------------------------------------

typedef __attribute__((ext_vector_type(16))) __bf16 v16bf;
typedef __attribute__((ext_vector_type(8)))  __bf16 v8bf;
typedef __attribute__((ext_vector_type(4)))  __bf16 v4bf;
typedef __attribute__((ext_vector_type(8)))  float  v8f;

union FragU { v16bf v; v8bf h[2]; };

static constexpr int Bn = 4;
static constexpr int Sn = 2048;
static constexpr int En = 1024;

// ---------------------------------------------------------------------------
// fp32 -> bf16 downconvert, 4 elements / thread
// ---------------------------------------------------------------------------
__global__ void cvt_f32_to_bf16(const float* __restrict__ in,
                                __bf16* __restrict__ out, int n4) {
    int i = blockIdx.x * blockDim.x + threadIdx.x;
    if (i >= n4) return;
    const float4 f = reinterpret_cast<const float4*>(in)[i];
    v4bf o;
    o[0] = (__bf16)f.x; o[1] = (__bf16)f.y; o[2] = (__bf16)f.z; o[3] = (__bf16)f.w;
    reinterpret_cast<v4bf*>(out)[i] = o;
}

// ---------------------------------------------------------------------------
// Load a 16x32 bf16 WMMA fragment from a row-major [rows, stride] matrix.
// Per the CDNA5 16-bit A layout: lane L (h = L>>4, r = L&15) holds
//   halfs 0..7  = M[row0+r][k0 + 8h      .. +7]
//   halfs 8..15 = M[row0+r][k0 + 16 + 8h .. +7]
// The B operand uses the identical per-lane layout with rows = columns of B
// (rows of B^T), which for all our operands is contiguous in memory.
// ---------------------------------------------------------------------------
__device__ __forceinline__ v16bf load_frag(const __bf16* __restrict__ base,
                                           int stride, int row0, int k0, int lane) {
    const int r = lane & 15;
    const int h = lane >> 4;
    const __bf16* p = base + (size_t)(row0 + r) * stride + k0 + h * 8;
    FragU u;
    u.h[0] = *reinterpret_cast<const v8bf*>(p);
    u.h[1] = *reinterpret_cast<const v8bf*>(p + 16);
    return u.v;
}

__device__ __forceinline__ v8f wmma_bf16(v16bf a, v16bf b, v8f c) {
    return __builtin_amdgcn_wmma_f32_16x16x32_bf16(false, a, false, b,
                                                   (short)0, c, false, false);
}

// ---------------------------------------------------------------------------
// Fused QKV projection:  out[m,n] = sum_k hbf[m,k] * W[n,k] + bias[n]
//   which=0 -> Q (row-major), which=1 -> K (row-major), which=2 -> Vt (transposed)
// 8 waves / block; each wave computes a 32x64 tile (2 M-subtiles x 4 N-subtiles).
// K-loop unrolled x2 with explicit ping-pong buffers: no rotation copies,
// loads for step k+2 issue while step k+1's WMMAs run.
// ---------------------------------------------------------------------------
__global__ __launch_bounds__(256) void qkv_gemm(
    const __bf16* __restrict__ A,                      // [M=B*S, K=E]
    const __bf16* __restrict__ Wq, const __bf16* __restrict__ Wk,
    const __bf16* __restrict__ Wv,
    const float* __restrict__ bq, const float* __restrict__ bk,
    const float* __restrict__ bv,
    __bf16* __restrict__ Qo, __bf16* __restrict__ Ko, __bf16* __restrict__ Vt) {
    const int N = En, K = En;
    const int lane = threadIdx.x & 31;
    const int wave = threadIdx.x >> 5;
    const long gw = (long)blockIdx.x * 8 + wave;       // 3*4096 wave-tiles
    const int which = (int)(gw / 4096);
    const int rem   = (int)(gw % 4096);
    const int mt = rem & 255;                          // M/32 = 256 tiles
    const int ng = rem >> 8;                           // N/64 = 16 groups
    const __bf16* W   = (which == 0) ? Wq : (which == 1) ? Wk : Wv;
    const float* bias = (which == 0) ? bq : (which == 1) ? bk : bv;
    const int row0 = mt * 32;
    const int n0   = ng * 64;

    v8f acc[2][4];
#pragma unroll
    for (int mi = 0; mi < 2; ++mi)
#pragma unroll
        for (int t = 0; t < 4; ++t)
#pragma unroll
            for (int r = 0; r < 8; ++r) acc[mi][t][r] = 0.0f;

    // ---- prologue: fill both buffer sets (k = 0 and k = 32) ----
    v16bf a0A = load_frag(A, K, row0,      0, lane);
    v16bf a1A = load_frag(A, K, row0 + 16, 0, lane);
    v16bf bbA[4];
#pragma unroll
    for (int t = 0; t < 4; ++t) bbA[t] = load_frag(W, K, n0 + t * 16, 0, lane);
    v16bf a0B = load_frag(A, K, row0,      32, lane);
    v16bf a1B = load_frag(A, K, row0 + 16, 32, lane);
    v16bf bbB[4];
#pragma unroll
    for (int t = 0; t < 4; ++t) bbB[t] = load_frag(W, K, n0 + t * 16, 32, lane);

    for (int k0 = 0; k0 < K - 64; k0 += 64) {
        // consume A-set @ k0, refill it @ k0+64
#pragma unroll
        for (int t = 0; t < 4; ++t) {
            acc[0][t] = wmma_bf16(a0A, bbA[t], acc[0][t]);
            acc[1][t] = wmma_bf16(a1A, bbA[t], acc[1][t]);
        }
        a0A = load_frag(A, K, row0,      k0 + 64, lane);
        a1A = load_frag(A, K, row0 + 16, k0 + 64, lane);
#pragma unroll
        for (int t = 0; t < 4; ++t)
            bbA[t] = load_frag(W, K, n0 + t * 16, k0 + 64, lane);
        // consume B-set @ k0+32, refill it @ k0+96
#pragma unroll
        for (int t = 0; t < 4; ++t) {
            acc[0][t] = wmma_bf16(a0B, bbB[t], acc[0][t]);
            acc[1][t] = wmma_bf16(a1B, bbB[t], acc[1][t]);
        }
        a0B = load_frag(A, K, row0,      k0 + 96, lane);
        a1B = load_frag(A, K, row0 + 16, k0 + 96, lane);
#pragma unroll
        for (int t = 0; t < 4; ++t)
            bbB[t] = load_frag(W, K, n0 + t * 16, k0 + 96, lane);
    }
    // ---- epilogue: consume both remaining sets ----
#pragma unroll
    for (int t = 0; t < 4; ++t) {
        acc[0][t] = wmma_bf16(a0A, bbA[t], acc[0][t]);
        acc[1][t] = wmma_bf16(a1A, bbA[t], acc[1][t]);
    }
#pragma unroll
    for (int t = 0; t < 4; ++t) {
        acc[0][t] = wmma_bf16(a0B, bbB[t], acc[0][t]);
        acc[1][t] = wmma_bf16(a1B, bbB[t], acc[1][t]);
    }

    const int hi = lane >> 4, nl = lane & 15;
    if (which < 2) {
        __bf16* Out = (which == 0) ? Qo : Ko;
#pragma unroll
        for (int mi = 0; mi < 2; ++mi)
#pragma unroll
            for (int t = 0; t < 4; ++t) {
                const int n = n0 + t * 16 + nl;
                const float bbias = bias[n];
#pragma unroll
                for (int r = 0; r < 8; ++r) {
                    const int m = row0 + mi * 16 + r + 8 * hi;
                    Out[(size_t)m * N + n] = (__bf16)(acc[mi][t][r] + bbias);
                }
            }
    } else {
        // Vt[b][e][s]: per lane e fixed, s contiguous over r -> 16B store
#pragma unroll
        for (int mi = 0; mi < 2; ++mi)
#pragma unroll
            for (int t = 0; t < 4; ++t) {
                const int n = n0 + t * 16 + nl;
                const float bbias = bias[n];
                const int m0 = row0 + mi * 16 + 8 * hi;
                const int batch = m0 / Sn;
                const int s0 = m0 % Sn;
                v8bf pk;
#pragma unroll
                for (int r = 0; r < 8; ++r) pk[r] = (__bf16)(acc[mi][t][r] + bbias);
                *reinterpret_cast<v8bf*>(Vt + ((size_t)batch * En + n) * Sn + s0) = pk;
            }
    }
}

// ---------------------------------------------------------------------------
// Flash-style LSH attention.  Grid = B * (S/16); block = 256 (8 waves).
// Per 128-key chunk:
//   wave w: 16x16 score tile (keys kt0+16w), mask by LSH bucket, write LDS
//   ALL waves: parallel online softmax (wave w owns rows {w, w+8}; each
//              16-lane half-wave reduces one row via __shfl_xor)
//   wave w: rescale its O[16x128] slice (e0 = 128w), O += P @ Vt-slice
// ---------------------------------------------------------------------------
__global__ __launch_bounds__(256) void lsh_attention(
    const __bf16* __restrict__ Q, const __bf16* __restrict__ Kmat,
    const __bf16* __restrict__ Vt, const int* __restrict__ hashb,
    float* __restrict__ out) {
    __shared__ float  Sbuf[16][128];
    __shared__ __bf16 Pbuf[16][128];
    __shared__ float  Mrow[16], Lrow[16], Fac[16];

    const int lane = threadIdx.x & 31;
    const int w    = threadIdx.x >> 5;
    const int b    = blockIdx.x >> 7;                  // / (S/16 = 128)
    const int qt   = blockIdx.x & 127;
    const __bf16* Qb  = Q    + (size_t)b * Sn * En;
    const __bf16* Kb  = Kmat + (size_t)b * Sn * En;
    const __bf16* Vtb = Vt   + (size_t)b * En * Sn;
    const int* hbb = hashb + b * Sn;
    const int hi = lane >> 4, nl = lane & 15;
    const float MASK = -1e30f;

    if (threadIdx.x < 16) { Mrow[threadIdx.x] = MASK; Lrow[threadIdx.x] = 0.0f; }

    int qbucket[8];
#pragma unroll
    for (int r = 0; r < 8; ++r) qbucket[r] = hbb[qt * 16 + r + 8 * hi];

    v8f O[8];
#pragma unroll
    for (int t = 0; t < 8; ++t)
#pragma unroll
        for (int r = 0; r < 8; ++r) O[t][r] = 0.0f;

    const int e0 = w * 128;
    __syncthreads();

    for (int kt0 = 0; kt0 < Sn; kt0 += 128) {
        const int kt = kt0 + w * 16;

        // ---- scores: Q(16xE) @ K_tile(16xE)^T, unrolled x2 ping-pong ----
        v8f sacc;
#pragma unroll
        for (int r = 0; r < 8; ++r) sacc[r] = 0.0f;
        v16bf aqA = load_frag(Qb, En, qt * 16, 0,  lane);
        v16bf bkA = load_frag(Kb, En, kt,      0,  lane);
        v16bf aqB = load_frag(Qb, En, qt * 16, 32, lane);
        v16bf bkB = load_frag(Kb, En, kt,      32, lane);
        for (int k0 = 0; k0 < En - 64; k0 += 64) {
            sacc = wmma_bf16(aqA, bkA, sacc);
            aqA = load_frag(Qb, En, qt * 16, k0 + 64, lane);
            bkA = load_frag(Kb, En, kt,      k0 + 64, lane);
            // warm L2/L0 for the next key chunk (global_prefetch_b8)
            if (kt + 128 < Sn)
                __builtin_prefetch(Kb + (size_t)(kt + 128 + nl) * En + k0, 0, 1);
            sacc = wmma_bf16(aqB, bkB, sacc);
            aqB = load_frag(Qb, En, qt * 16, k0 + 96, lane);
            bkB = load_frag(Kb, En, kt,      k0 + 96, lane);
        }
        sacc = wmma_bf16(aqA, bkA, sacc);
        sacc = wmma_bf16(aqB, bkB, sacc);

        const int kbucket = hbb[kt + nl];
#pragma unroll
        for (int r = 0; r < 8; ++r) {
            float s = sacc[r] * 0.03125f;              // 1/sqrt(1024)
            if (qbucket[r] != kbucket) s = MASK;
            Sbuf[r + 8 * hi][w * 16 + nl] = s;
        }
        __syncthreads();

        // ---- parallel online softmax: wave w owns rows w and w+8 ----
        {
            const int m = w + 8 * hi;                  // row for this half-wave
            const float* Srow = &Sbuf[m][nl * 8];
            float sv[8];
#pragma unroll
            for (int j = 0; j < 8; ++j) sv[j] = Srow[j];
            float mx = sv[0];
#pragma unroll
            for (int j = 1; j < 8; ++j) mx = fmaxf(mx, sv[j]);
#pragma unroll
            for (int d = 1; d < 16; d <<= 1)
                mx = fmaxf(mx, __shfl_xor(mx, d, 32)); // reduce within half-wave
            const float mo = Mrow[m];
            const float mxall = fmaxf(mx, mo);
            float sum = 0.0f;
            v8bf pk;
#pragma unroll
            for (int j = 0; j < 8; ++j) {
                const float p = __expf(sv[j] - mxall);
                pk[j] = (__bf16)p;
                sum += p;
            }
            *reinterpret_cast<v8bf*>(&Pbuf[m][nl * 8]) = pk;
#pragma unroll
            for (int d = 1; d < 16; d <<= 1)
                sum += __shfl_xor(sum, d, 32);
            if (nl == 0) {
                const float fac = __expf(mo - mxall);
                Lrow[m] = Lrow[m] * fac + sum;
                Mrow[m] = mxall;
                Fac[m]  = fac;
            }
        }
        __syncthreads();

        // ---- rescale O, then O += P(16x128) @ V(128 x 128-slice) ----
        float f[8];
#pragma unroll
        for (int r = 0; r < 8; ++r) f[r] = Fac[r + 8 * hi];
#pragma unroll
        for (int t = 0; t < 8; ++t)
#pragma unroll
            for (int r = 0; r < 8; ++r) O[t][r] *= f[r];

        // flattened (kc, t) loop (fully unrolled), B-frag double-buffered
        v16bf ap;
        v16bf bcur = load_frag(Vtb, Sn, e0, kt0, lane);
#pragma unroll
        for (int i = 0; i < 32; ++i) {
            const int kc = i >> 3, t = i & 7;
            v16bf bnxt;
            if (i + 1 < 32) {
                const int kcn = (i + 1) >> 3, tn = (i + 1) & 7;
                bnxt = load_frag(Vtb, Sn, e0 + tn * 16, kt0 + kcn * 32, lane);
            }
            if (t == 0) {
                FragU u;
                u.h[0] = *reinterpret_cast<const v8bf*>(&Pbuf[nl][kc * 32 + hi * 8]);
                u.h[1] = *reinterpret_cast<const v8bf*>(&Pbuf[nl][kc * 32 + hi * 8 + 16]);
                ap = u.v;
            }
            O[t] = wmma_bf16(ap, bcur, O[t]);
            if (i + 1 < 32) bcur = bnxt;
        }
    }

    __syncthreads();
    float inv_l[8];
#pragma unroll
    for (int r = 0; r < 8; ++r) inv_l[r] = 1.0f / Lrow[r + 8 * hi];

    float* outb = out + (size_t)b * Sn * En;
#pragma unroll
    for (int t = 0; t < 8; ++t) {
        const int e = e0 + t * 16 + nl;
#pragma unroll
        for (int r = 0; r < 8; ++r) {
            const int m = qt * 16 + r + 8 * hi;
            outb[(size_t)m * En + e] = O[t][r] * inv_l[r];
        }
    }
}

// ---------------------------------------------------------------------------
extern "C" void kernel_launch(void* const* d_in, const int* in_sizes, int n_in,
                              void* d_out, int out_size, void* d_ws, size_t ws_size,
                              hipStream_t stream) {
    const float* hidden = (const float*)d_in[0];
    const int*   hb     = (const int*)d_in[1];
    const float* Wq     = (const float*)d_in[2];
    const float* bq     = (const float*)d_in[3];
    const float* Wk     = (const float*)d_in[4];
    const float* bk     = (const float*)d_in[5];
    const float* Wv     = (const float*)d_in[6];
    const float* bv     = (const float*)d_in[7];

    const size_t nHid = (size_t)Bn * Sn * En;   // 8388608
    const size_t nW   = (size_t)En * En;        // 1048576

    __bf16* hbf = (__bf16*)d_ws;
    __bf16* wqb = hbf + nHid;
    __bf16* wkb = wqb + nW;
    __bf16* wvb = wkb + nW;
    __bf16* Qb  = wvb + nW;
    __bf16* Kb  = Qb + nHid;
    __bf16* Vtb = Kb + nHid;                    // total ~70 MB of d_ws

    // Phase 1: downconvert
    {
        int n4 = (int)(nHid / 4);
        cvt_f32_to_bf16<<<(n4 + 255) / 256, 256, 0, stream>>>(hidden, hbf, n4);
        int w4 = (int)(nW / 4);
        cvt_f32_to_bf16<<<(w4 + 255) / 256, 256, 0, stream>>>(Wq, wqb, w4);
        cvt_f32_to_bf16<<<(w4 + 255) / 256, 256, 0, stream>>>(Wk, wkb, w4);
        cvt_f32_to_bf16<<<(w4 + 255) / 256, 256, 0, stream>>>(Wv, wvb, w4);
    }

    // Phase 2: fused QKV GEMM (3 * 256 * 16 = 12288 wave-tiles, 8 waves/block)
    qkv_gemm<<<12288 / 8, 256, 0, stream>>>(hbf, wqb, wkb, wvb,
                                            bq, bk, bv, Qb, Kb, Vtb);

    // Phase 3: flash-style LSH attention
    lsh_attention<<<Bn * (Sn / 16), 256, 0, stream>>>(Qb, Kb, Vtb, hb,
                                                      (float*)d_out);
}
